// CrossAttention_9560597201230
// MI455X (gfx1250) — matrix operand order repaired and checked by reading, not compiled, not run on hardware
//
#include <hip/hip_runtime.h>

#define SEQ    2048
#define D_EMB  1024
#define NBATCH 4
#define PITCH  40   // LDS row pitch in ushorts (80B, 16B-aligned)

typedef __attribute__((ext_vector_type(16))) __bf16 v16bf;
typedef __attribute__((ext_vector_type(8)))  float  v8f;

__device__ __forceinline__ unsigned short f32_to_bf16(float f) {
  unsigned int u = __builtin_bit_cast(unsigned int, f);
  unsigned int r = u + 0x7FFFu + ((u >> 16) & 1u);   // round-to-nearest-even
  return (unsigned short)(r >> 16);
}

__device__ __forceinline__ unsigned int cvt2_bf16(float lo, float hi) {
#if __has_builtin(__builtin_amdgcn_cvt_pk_bf16_f32)
  typedef __attribute__((ext_vector_type(2))) __bf16 v2bf;
  v2bf p = __builtin_amdgcn_cvt_pk_bf16_f32(lo, hi);
  return __builtin_bit_cast(unsigned int, p);
#else
  return (unsigned)f32_to_bf16(lo) | ((unsigned)f32_to_bf16(hi) << 16);
#endif
}

// Async global->LDS copy of 16 bytes (CDNA5 async path, tracked by ASYNCcnt).
__device__ __forceinline__ void async_b128(unsigned lds_addr, unsigned goff,
                                           unsigned long long sbase) {
  asm volatile("global_load_async_to_lds_b128 %0, %1, %2"
               :: "v"(lds_addr), "v"(goff), "s"(sbase) : "memory");
}
__device__ __forceinline__ void wait_async0() {
  asm volatile("s_wait_asynccnt 0" ::: "memory");
}
__device__ __forceinline__ unsigned lds_off(const void* p) {
  return (unsigned)(unsigned long long)(size_t)p;  // low 32 bits = LDS byte offset
}

// ---- WMMA fragment loaders from LDS ---------------------------------------
// A-matrix 16x32 bf16 (M x K), ISA layout:
//   lanes 0-15:  row=lane,    dwords {0..3, 8..11}   (K 0-7, 16-23)
//   lanes 16-31: row=lane-16, dwords {4..7, 12..15}  (K 8-15, 24-31)
__device__ __forceinline__ v16bf frag_ld_a(const unsigned short* tile, int row_base, int lane) {
  const unsigned int* pd = (const unsigned int*)(tile + (size_t)(row_base + (lane & 15)) * PITCH);
  const int off = (lane & 16) ? 4 : 0;
  union { v16bf v; uint4 q[2]; } u;
  u.q[0] = *(const uint4*)(pd + off);
  u.q[1] = *(const uint4*)(pd + off + 8);
  return u.v;
}

// B-matrix 32x16 bf16 (K x N) staged transposed in LDS as Bt[n][k]:
//   lanes 0-15:  col=lane,    dwords {0..3, 8..11}   (K 0-7, 16-23)
//   lanes 16-31: col=lane-16, dwords {4..7, 12..15}  (K 8-15, 24-31)
__device__ __forceinline__ v16bf frag_ld_b(const unsigned short* tile, int row_base, int lane) {
  const unsigned int* pd = (const unsigned int*)(tile + (size_t)(row_base + (lane & 15)) * PITCH);
  const int off = (lane & 16) ? 4 : 0;
  union { v16bf v; uint4 q[2]; } u;
  u.q[0] = *(const uint4*)(pd + off);
  u.q[1] = *(const uint4*)(pd + off + 8);
  return u.v;
}

// Wave-level 32x64 MMA step over one K-chunk of 32.
__device__ __forceinline__ void wave_mma(const unsigned short* lA, const unsigned short* lB,
                                         int wm, int wn, int lane, v8f (&acc)[2][4]) {
  v16bf af[2], bfm[4];
#pragma unroll
  for (int i = 0; i < 2; ++i) af[i] = frag_ld_a(lA, wm + i * 16, lane);
#pragma unroll
  for (int j = 0; j < 4; ++j) bfm[j] = frag_ld_b(lB, wn + j * 16, lane);
#pragma unroll
  for (int i = 0; i < 2; ++i)
#pragma unroll
    for (int j = 0; j < 4; ++j)
      acc[i][j] = __builtin_amdgcn_wmma_f32_16x16x32_bf16(
          false, af[i], false, bfm[j], (short)0, acc[i][j], false, false);
}

__device__ __forceinline__ void acc_zero(v8f (&acc)[2][4]) {
#pragma unroll
  for (int i = 0; i < 2; ++i)
#pragma unroll
    for (int j = 0; j < 4; ++j)
#pragma unroll
      for (int e = 0; e < 8; ++e) acc[i][j][e] = 0.0f;
}

// ---- Stage 1: C[bf16] = A[f32] @ W[f32] ------------------------------------
// Single-barrier, double-buffered, register-staged pipeline:
//   stage i: store chunk i+1 regs->LDS[next], WMMA from LDS[cur], load chunk i+2.
__global__ __launch_bounds__(256) void k_gemm_qkv(const float* __restrict__ A,
                                                  const float* __restrict__ W,
                                                  unsigned short* __restrict__ C,
                                                  int N, int K) {
  __shared__ unsigned short lA[2][128 * PITCH];   // [m][k]
  __shared__ unsigned short lB[2][128 * PITCH];   // [n][k] (transposed W)
  const int t    = threadIdx.x;
  const int lane = t & 31;
  const int wave = t >> 5;
  const int wm   = (wave & 3) * 32;
  const int wn   = (wave >> 2) * 64;
  const int bm   = blockIdx.x * 128;
  const int bn   = blockIdx.y * 128;

  v8f acc[2][4];
  acc_zero(acc);

  const int arow  = t >> 1;
  const int ahalf = (t & 1) * 16;
  const int wk    = t >> 3;         // 0..31 (k row of W tile)
  const int wcol  = (t & 7) * 16;   // n sub-column

  float4 ar[4], wr[4];
  auto gload = [&](int k0) {
    const float4* asrc = (const float4*)(A + (size_t)(bm + arow) * K + k0 + ahalf);
#pragma unroll
    for (int i = 0; i < 4; ++i) ar[i] = asrc[i];
    const float4* wsrc = (const float4*)(W + (size_t)(k0 + wk) * N + bn + wcol);
#pragma unroll
    for (int i = 0; i < 4; ++i) wr[i] = wsrc[i];
  };
  auto lstore = [&](int buf) {
    unsigned int ta[8];
#pragma unroll
    for (int i = 0; i < 4; ++i) {
      ta[2 * i + 0] = cvt2_bf16(ar[i].x, ar[i].y);
      ta[2 * i + 1] = cvt2_bf16(ar[i].z, ar[i].w);
    }
    uint4* dst = (uint4*)&lA[buf][arow * PITCH + ahalf];
    dst[0] = *(const uint4*)&ta[0];
    dst[1] = *(const uint4*)&ta[4];
    unsigned short tw[16];
#pragma unroll
    for (int i = 0; i < 4; ++i) {
      tw[4 * i + 0] = f32_to_bf16(wr[i].x);
      tw[4 * i + 1] = f32_to_bf16(wr[i].y);
      tw[4 * i + 2] = f32_to_bf16(wr[i].z);
      tw[4 * i + 3] = f32_to_bf16(wr[i].w);
    }
#pragma unroll
    for (int i = 0; i < 16; ++i) lB[buf][(wcol + i) * PITCH + wk] = tw[i];
  };

  const int NS = K / 32;   // 32 stages
  gload(0);
  lstore(0);
  gload(32);
  __syncthreads();

  for (int i = 0; i < NS; ++i) {
    const int cur = i & 1;
    if (i + 1 < NS) lstore(cur ^ 1);       // chunk i+1 into alternate buffer
    wave_mma(lA[cur], lB[cur], wm, wn, lane, acc);
    if (i + 2 < NS) gload((i + 2) * 32);   // prefetch chunk i+2 into regs
    __syncthreads();
  }

  const int cl = lane & 15;
  const int rh = (lane >> 4) * 8;
#pragma unroll
  for (int i = 0; i < 2; ++i)
#pragma unroll
    for (int j = 0; j < 4; ++j)
#pragma unroll
      for (int e = 0; e < 8; ++e) {
        int r = bm + wm + i * 16 + rh + e;
        int c = bn + wn + j * 16 + cl;
        C[(size_t)r * N + c] = f32_to_bf16(acc[i][j][e]);
      }
}

// ---- Stage 2: S[f32] = scale * Q @ K^T  (double-buffered async copies) -----
__global__ __launch_bounds__(256) void k_scores(const unsigned short* __restrict__ Qb,
                                                const unsigned short* __restrict__ KVb,
                                                float* __restrict__ S, float scale) {
  __shared__ unsigned short lA[2][128 * PITCH];
  __shared__ unsigned short lB[2][128 * PITCH];
  const int t    = threadIdx.x;
  const int lane = t & 31;
  const int wave = t >> 5;
  const int wm   = (wave & 3) * 32;
  const int wn   = (wave >> 2) * 64;
  const int bq   = blockIdx.x * 128;
  const int bk   = blockIdx.y * 128;
  const int b    = blockIdx.z;
  const unsigned short* Q0 = Qb + (size_t)b * SEQ * D_EMB;
  const unsigned short* K0 = KVb + (size_t)b * SEQ * (2 * D_EMB);  // K = cols 0..1023

  v8f acc[2][4];
  acc_zero(acc);

  const int row  = t >> 1;
  const int part = (t & 1) * 16;
  const unsigned long long qbase = (unsigned long long)(size_t)Q0;
  const unsigned long long kbase = (unsigned long long)(size_t)K0;
  const unsigned qrow = (unsigned)((bq + row) * D_EMB + part) * 2u;        // bytes
  const unsigned krow = (unsigned)((bk + row) * (2 * D_EMB) + part) * 2u;  // bytes

  auto issue = [&](int buf, int d0) {
    unsigned la = lds_off(&lA[buf][row * PITCH + part]);
    unsigned lb = lds_off(&lB[buf][row * PITCH + part]);
    unsigned qo = qrow + (unsigned)(d0 * 2);
    unsigned ko = krow + (unsigned)(d0 * 2);
    async_b128(la,      qo,      qbase);
    async_b128(la + 16, qo + 16, qbase);
    async_b128(lb,      ko,      kbase);
    async_b128(lb + 16, ko + 16, kbase);
  };

  issue(0, 0);
  wait_async0();
  __syncthreads();

  for (int i = 0; i < D_EMB / 32; ++i) {
    const int cur = i & 1;
    if (i + 1 < D_EMB / 32) issue(cur ^ 1, (i + 1) * 32);   // DMA overlaps WMMA
    wave_mma(lA[cur], lB[cur], wm, wn, lane, acc);
    wait_async0();
    __syncthreads();
  }

  float* Sb = S + (size_t)b * SEQ * SEQ;
  const int cl = lane & 15;
  const int rh = (lane >> 4) * 8;
#pragma unroll
  for (int i = 0; i < 2; ++i)
#pragma unroll
    for (int j = 0; j < 4; ++j)
#pragma unroll
      for (int e = 0; e < 8; ++e) {
        int r = bq + wm + i * 16 + rh + e;
        int c = bk + wn + j * 16 + cl;
        Sb[(size_t)r * SEQ + c] = acc[i][j][e] * scale;
      }
}

// ---- Stage 3: in-place row softmax; writes bf16 P over the f32 row ---------
// Vectorized: 2x float4 loads, 1x uint4 (8 x bf16) store per thread.
__global__ __launch_bounds__(256) void k_softmax(float* __restrict__ S) {
  const int r = blockIdx.x;  // 0 .. NBATCH*SEQ-1
  const int t = threadIdx.x;
  const float4* rp = (const float4*)(S + (size_t)r * SEQ) + 2 * t;
  __shared__ float red[256];

  float4 a = rp[0];
  float4 b = rp[1];
  float m = fmaxf(fmaxf(fmaxf(a.x, a.y), fmaxf(a.z, a.w)),
                  fmaxf(fmaxf(b.x, b.y), fmaxf(b.z, b.w)));
  red[t] = m;
  __syncthreads();
  for (int s = 128; s > 0; s >>= 1) {
    if (t < s) red[t] = fmaxf(red[t], red[t + s]);
    __syncthreads();
  }
  m = red[0];
  __syncthreads();

  a.x = __expf(a.x - m); a.y = __expf(a.y - m);
  a.z = __expf(a.z - m); a.w = __expf(a.w - m);
  b.x = __expf(b.x - m); b.y = __expf(b.y - m);
  b.z = __expf(b.z - m); b.w = __expf(b.w - m);
  float sum = (a.x + a.y) + (a.z + a.w) + (b.x + b.y) + (b.z + b.w);
  red[t] = sum;
  __syncthreads();
  for (int s = 128; s > 0; s >>= 1) {
    if (t < s) red[t] += red[t + s];
    __syncthreads();
  }
  const float inv = 1.0f / red[0];

  uint4 pk;
  pk.x = cvt2_bf16(a.x * inv, a.y * inv);
  pk.y = cvt2_bf16(a.z * inv, a.w * inv);
  pk.z = cvt2_bf16(b.x * inv, b.y * inv);
  pk.w = cvt2_bf16(b.z * inv, b.w * inv);
  uint4* prow = (uint4*)((unsigned short*)S + (size_t)r * 4096);
  prow[t] = pk;
}

// ---- Stage 4: O[f32] = P[bf16] @ V[bf16] (async P, reg-staged V transpose) -
__global__ __launch_bounds__(256) void k_pv(const float* __restrict__ Sbuf,
                                            const unsigned short* __restrict__ KVb,
                                            float* __restrict__ O) {
  __shared__ unsigned short lA[2][128 * PITCH];
  __shared__ unsigned short lB[2][128 * PITCH];
  const int t    = threadIdx.x;
  const int lane = t & 31;
  const int wave = t >> 5;
  const int wm   = (wave & 3) * 32;
  const int wn   = (wave >> 2) * 64;
  const int bq   = blockIdx.x * 128;
  const int bd   = blockIdx.y * 128;
  const int b    = blockIdx.z;
  const unsigned short* P0 = (const unsigned short*)Sbuf + (size_t)b * SEQ * 4096;
  const unsigned short* V0 = KVb + (size_t)b * SEQ * (2 * D_EMB) + D_EMB;  // V = cols 1024..2047

  v8f acc[2][4];
  acc_zero(acc);

  const int row  = t >> 1;
  const int part = (t & 1) * 16;
  const int vk   = t >> 3;         // 0..31 (kl row of V tile)
  const int vcol = (t & 7) * 16;   // d sub-column
  const unsigned long long pbase = (unsigned long long)(size_t)P0;
  const unsigned prow = (unsigned)(bq + row) * 8192u + (unsigned)part * 2u;  // bytes

  auto issueP = [&](int buf, int c0) {
    unsigned la = lds_off(&lA[buf][row * PITCH + part]);
    unsigned po = prow + (unsigned)(c0 * 2);
    async_b128(la,      po,      pbase);
    async_b128(la + 16, po + 16, pbase);
  };

  uint4 vr0, vr1;                      // register-staged V chunk (16 bf16)
  auto vload = [&](int c0) {
    const uint4* s = (const uint4*)(V0 + (size_t)(c0 + vk) * (2 * D_EMB) + bd + vcol);
    vr0 = s[0];
    vr1 = s[1];
  };
  auto vstore = [&](int buf) {
    unsigned short tmp[16];
    *(uint4*)&tmp[0] = vr0;
    *(uint4*)&tmp[8] = vr1;
#pragma unroll
    for (int i = 0; i < 16; ++i) lB[buf][(vcol + i) * PITCH + vk] = tmp[i];
  };

  vload(0);
  issueP(0, 0);
  vstore(0);
  wait_async0();
  __syncthreads();

  for (int i = 0; i < SEQ / 32; ++i) {
    const int cur = i & 1;
    if (i + 1 < SEQ / 32) {
      issueP(cur ^ 1, (i + 1) * 32);   // async DMA overlaps with WMMA below
      vload((i + 1) * 32);
    }
    wave_mma(lA[cur], lB[cur], wm, wn, lane, acc);
    if (i + 1 < SEQ / 32) vstore(cur ^ 1);
    wait_async0();
    __syncthreads();
  }

  float* Ob = O + (size_t)b * SEQ * D_EMB;
  const int cl = lane & 15;
  const int rh = (lane >> 4) * 8;
#pragma unroll
  for (int i = 0; i < 2; ++i)
#pragma unroll
    for (int j = 0; j < 4; ++j)
#pragma unroll
      for (int e = 0; e < 8; ++e) {
        int r = bq + wm + i * 16 + rh + e;
        int c = bd + wn + j * 16 + cl;
        Ob[(size_t)r * D_EMB + c] = acc[i][j][e];
      }
}

// ---- Launch ----------------------------------------------------------------
extern "C" void kernel_launch(void* const* d_in, const int* in_sizes, int n_in,
                              void* d_out, int out_size, void* d_ws, size_t ws_size,
                              hipStream_t stream) {
  const float* H   = (const float*)d_in[0];  // [4, 2048, 1024]
  const float* DH  = (const float*)d_in[1];  // [4, 2048, 1024]
  const float* Wq  = (const float*)d_in[2];  // [1024, 1024]
  const float* Wkv = (const float*)d_in[3];  // [1024, 2048]
  float* out = (float*)d_out;                // [4, 2048, 1024]

  char* ws = (char*)d_ws;
  unsigned short* Qb  = (unsigned short*)ws;                                  // 16 MB
  unsigned short* KVb = (unsigned short*)(ws + (size_t)16 * 1024 * 1024);     // 32 MB
  float*          S   = (float*)(ws + (size_t)48 * 1024 * 1024);              // 64 MB (P aliased in-place)

  const dim3 blk(256);
  const int M = NBATCH * SEQ;  // 8192

  // Q = H @ Wq
  k_gemm_qkv<<<dim3(M / 128, D_EMB / 128), blk, 0, stream>>>(H, Wq, Qb, D_EMB, D_EMB);
  // KV = DH @ Wkv
  k_gemm_qkv<<<dim3(M / 128, (2 * D_EMB) / 128), blk, 0, stream>>>(DH, Wkv, KVb, 2 * D_EMB, D_EMB);
  // S = scale * Q K^T
  k_scores<<<dim3(SEQ / 128, SEQ / 128, NBATCH), blk, 0, stream>>>(Qb, KVb, S, 1.0f / 32.0f);
  // softmax rows (in-place, writes bf16 P)
  k_softmax<<<dim3(M), blk, 0, stream>>>(S);
  // O = P @ V
  k_pv<<<dim3(SEQ / 128, D_EMB / 128, NBATCH), blk, 0, stream>>>(S, KVb, out);
}
